// RaceInteractionBlock_55370718380449
// MI455X (gfx1250) — matrix operand order repaired and checked by reading, not compiled, not run on hardware
//
#include <hip/hip_runtime.h>

#define NN 10000
#define EE 160000
#define FF 128
#define SS 10
#define RR 8
#define HH 64

typedef __attribute__((ext_vector_type(16))) _Float16 v16h;
typedef __attribute__((ext_vector_type(8)))  float    v8f;

// scale factors from the reference
#define S_SKIP 0.027950849718747373f   // 1/sqrt(128*10)
#define S_H    0.08838834764831845f    // 1/sqrt(128)
#define S_M1   0.35355339059327373f    // 1/sqrt(8)
#define S_M2   0.125f                  // 1/sqrt(64)
#define S_W4   0.125f                  // 1/sqrt(64)
#define S_X1   0.022097086912079608f   // (1/sqrt(16)) / sqrt(128)
#define S_X2   S_SKIP
#define S_OUT  S_H

// f16 swizzled-weight arena offsets (in halves)
#define OFF_WUP  0        // 128x128
#define OFF_W1   16384    // 32x64 (zero-padded K: real K=8)
#define OFF_W2   18432    // 64x64
#define OFF_W3   22528    // 64x64
#define OFF_W4   26624    // 64x128 (first F cols of mlp_w4)
#define OFF_WLIN 34816    // 128x128 (W_lin[0])
#define OFF_WOUT 51200    // 128x128
#define OFF_WSN  67584    // 10 x 128x128 (W_skip_node per species)
#define OFF_WSM  231424   // 10 x 128x128 (W_skip_msg[0] per species)
#define WF_TOTAL 395264

__device__ __forceinline__ float silu(float x) {
  return x * (1.0f / (1.0f + __expf(-x)));
}

__device__ __forceinline__ v8f wmma_f16(v16h a, v16h b, v8f c) {
  // D = A(16x32 f16) * B(32x16 f16) + C(16x16 f32)
  return __builtin_amdgcn_wmma_f32_16x16x32_f16(false, a, false, b, (short)0, c,
                                                false, false);
}

// A fragment (16x32, f16) from a row-major f16 buffer with leading dim `ld`.
// Lane layout per ISA: lanes 0-15 halves0-7 -> K=0..7, halves8-15 -> K=16..23;
// lanes 16-31: K=8..15 / K=24..31. Row M = lane&15.
__device__ __forceinline__ v16h a_frag(const _Float16* base, int ld, int koff) {
  int lane = threadIdx.x & 31;
  const _Float16* p = base + (lane & 15) * ld + koff + ((lane >> 4) << 3);
  v16h a;
#pragma unroll
  for (int j = 0; j < 8; ++j) { a[j] = p[j]; a[8 + j] = p[16 + j]; }
  return a;
}

// B fragment from the pre-swizzled weight arena: each lane's 16 halves are
// contiguous (32B) -> coalesced b128 loads. ntn = Ncols/16.
__device__ __forceinline__ v16h b_frag(const _Float16* Wsw, int ntn, int kt, int nt) {
  int lane = threadIdx.x & 31;
  return *(const v16h*)(Wsw + (size_t)((((kt * ntn + nt) << 5) + lane) << 4));
}

// Store a 16x16 f32 C tile (scaled, optional silu) as f16 row-major into LDS.
// C layout: lane = column (l&15), vgpr r -> row r + 8*(l>>4).
__device__ __forceinline__ void store_tile_f16(_Float16* base, int ld, int nt,
                                               v8f c, float scale, bool act) {
  int lane = threadIdx.x & 31;
  int col = nt * 16 + (lane & 15);
  int r0 = (lane >> 4) << 3;
#pragma unroll
  for (int r = 0; r < 8; ++r) {
    float v = c[r] * scale;
    if (act) v = silu(v);
    base[(r0 + r) * ld + col] = (_Float16)v;
  }
}

__device__ __forceinline__ void decode_kn(int li, int ntn, int& k, int& n) {
  int h = li & 15;
  int l = (li >> 4) & 31;
  int tile = li >> 9;
  int kt = tile / ntn, nt = tile - kt * ntn;
  k = kt * 32 + ((l >> 4) << 3) + (h & 7) + ((h >> 3) << 4);
  n = nt * 16 + (l & 15);
}

// ---------------- weight prep: f32 -> f16, B-fragment swizzle ----------------
__global__ void k_prep(const float* __restrict__ W_up, const float* __restrict__ w1,
                       const float* __restrict__ w2, const float* __restrict__ w3,
                       const float* __restrict__ w4, const float* __restrict__ wlin,
                       const float* __restrict__ wout, const float* __restrict__ wsn,
                       const float* __restrict__ wsm, _Float16* __restrict__ wf) {
  for (int i = blockIdx.x * blockDim.x + threadIdx.x; i < WF_TOTAL;
       i += gridDim.x * blockDim.x) {
    int k, n; float v;
    if (i < OFF_W1)        { decode_kn(i, 8, k, n);            v = W_up[k * FF + n]; }
    else if (i < OFF_W2)   { decode_kn(i - OFF_W1, 4, k, n);   v = (k < RR) ? w1[k * HH + n] : 0.0f; }
    else if (i < OFF_W3)   { decode_kn(i - OFF_W2, 4, k, n);   v = w2[k * HH + n]; }
    else if (i < OFF_W4)   { decode_kn(i - OFF_W3, 4, k, n);   v = w3[k * HH + n]; }
    else if (i < OFF_WLIN) { decode_kn(i - OFF_W4, 8, k, n);   v = w4[k * 4 * FF + n]; }
    else if (i < OFF_WOUT) { decode_kn(i - OFF_WLIN, 8, k, n); v = wlin[k * FF + n]; }
    else if (i < OFF_WSN)  { decode_kn(i - OFF_WOUT, 8, k, n); v = wout[k * FF + n]; }
    else if (i < OFF_WSM)  { int li = i - OFF_WSN; int s = li >> 14;
                             decode_kn(li & 16383, 8, k, n);   v = wsn[(k * SS + s) * FF + n]; }
    else                   { int li = i - OFF_WSM; int s = li >> 14;
                             decode_kn(li & 16383, 8, k, n);   v = wsm[(k * SS + s) * FF + n]; }
    wf[i] = (_Float16)v;
  }
}

__global__ void k_zero(float* __restrict__ p, int n) {
  for (int i = blockIdx.x * blockDim.x + threadIdx.x; i < n;
       i += gridDim.x * blockDim.x) p[i] = 0.0f;
}

// ---------------- node kernel: h = nf@W_up, skip = species-GEMM ----------------
__global__ __launch_bounds__(256) void k_node(const float* __restrict__ node_feats,
                                              const int* __restrict__ species,
                                              const _Float16* __restrict__ wf,
                                              float* __restrict__ h,
                                              float* __restrict__ skip_out) {
  __shared__ _Float16 lsA[8][16 * 136];   // padded stride (136) -> no bank conflicts
  __shared__ int lsp[8][16];
  int wave = threadIdx.x >> 5, lane = threadIdx.x & 31;
  int nb = (blockIdx.x * 8 + wave) * 16;
  if (nb >= NN) return;
  _Float16* A = lsA[wave];
  for (int i = lane; i < 16 * 128; i += 32) {
    int r = i >> 7, c = i & 127;
    A[r * 136 + c] = (_Float16)node_feats[(nb + r) * FF + c];
  }
  if (lane < 16) lsp[wave][lane] = species[nb + lane];
  __builtin_amdgcn_wave_barrier();

  int M = lane & 15, hi = lane >> 4;
  v16h af[4];
#pragma unroll
  for (int kf = 0; kf < 4; ++kf) af[kf] = a_frag(A, 136, kf * 32);

  // h = node_feats @ W_up * S_H
#pragma unroll
  for (int nt = 0; nt < 8; ++nt) {
    v8f c = {};
#pragma unroll
    for (int kf = 0; kf < 4; ++kf)
      c = wmma_f16(af[kf], b_frag(wf + OFF_WUP, 8, kf, nt), c);
#pragma unroll
    for (int r = 0; r < 8; ++r)
      h[(nb + hi * 8 + r) * FF + nt * 16 + M] = c[r] * S_H;
  }

  // skip: sum over species of (A masked by species) @ W_skip_node[:,s,:]
  int mysp = lsp[wave][M];
  v16h zv = {};
  v8f cs[8];
#pragma unroll
  for (int nt = 0; nt < 8; ++nt) { v8f z = {}; cs[nt] = z; }
  for (int s = 0; s < SS; ++s) {
    bool keep = (mysp == s);
    v16h az[4];
#pragma unroll
    for (int kf = 0; kf < 4; ++kf) az[kf] = keep ? af[kf] : zv;
    const _Float16* Ws = wf + OFF_WSN + s * (FF * FF);
#pragma unroll
    for (int nt = 0; nt < 8; ++nt)
#pragma unroll
      for (int kf = 0; kf < 4; ++kf)
        cs[nt] = wmma_f16(az[kf], b_frag(Ws, 8, kf, nt), cs[nt]);
  }
#pragma unroll
  for (int nt = 0; nt < 8; ++nt)
#pragma unroll
    for (int r = 0; r < 8; ++r)
      skip_out[(nb + hi * 8 + r) * FF + nt * 16 + M] = cs[nt][r] * S_SKIP;
}

// -------- edge kernel: fused MLP (WMMA) + gather h[sender] + scatter-add m --------
__global__ __launch_bounds__(256) void k_edge(const float* __restrict__ edge_feats,
                                              const float* __restrict__ edge_attrs,
                                              const int* __restrict__ eidx,
                                              const float* __restrict__ h,
                                              const _Float16* __restrict__ wf,
                                              float* __restrict__ m) {
  __shared__ _Float16 lsT[8][16 * 72];    // per-wave 16x64 activations, padded
  __shared__ int lsS[8][16], lsR[8][16];
  __shared__ float lsW[8][16];
  int wave = threadIdx.x >> 5, lane = threadIdx.x & 31;
  int eb = (blockIdx.x * 8 + wave) * 16;
  if (eb >= EE) return;
  if (lane < 16) {
    int e = eb + lane;
    lsS[wave][lane] = eidx[e];
    lsR[wave][lane] = eidx[EE + e];
    lsW[wave][lane] = edge_attrs[e * 16];   // sh_0 (l=0, d=1)
  }
  int M = lane & 15, hi = lane >> 4;

  // A1: 16 edges x K32 (real K=8, rest zero). Only lanes 0-15 carry K<8.
  v16h a1 = {};
  if (hi == 0) {
    const float* p = edge_feats + (eb + M) * RR;
#pragma unroll
    for (int j = 0; j < 8; ++j) a1[j] = (_Float16)p[j];
  }
  _Float16* T = lsT[wave];

  // L1: (16x8) @ (8x64), silu
#pragma unroll
  for (int nt = 0; nt < 4; ++nt) {
    v8f c = {};
    c = wmma_f16(a1, b_frag(wf + OFF_W1, 4, 0, nt), c);
    store_tile_f16(T, 72, nt, c, S_M1, true);
  }
  __builtin_amdgcn_wave_barrier();
  // L2: 64x64, silu
  {
    v16h x0 = a_frag(T, 72, 0), x1 = a_frag(T, 72, 32);
    __builtin_amdgcn_wave_barrier();
#pragma unroll
    for (int nt = 0; nt < 4; ++nt) {
      v8f c = {};
      c = wmma_f16(x0, b_frag(wf + OFF_W2, 4, 0, nt), c);
      c = wmma_f16(x1, b_frag(wf + OFF_W2, 4, 1, nt), c);
      store_tile_f16(T, 72, nt, c, S_M2, true);
    }
  }
  __builtin_amdgcn_wave_barrier();
  // L3: 64x64, silu
  {
    v16h x0 = a_frag(T, 72, 0), x1 = a_frag(T, 72, 32);
    __builtin_amdgcn_wave_barrier();
#pragma unroll
    for (int nt = 0; nt < 4; ++nt) {
      v8f c = {};
      c = wmma_f16(x0, b_frag(wf + OFF_W3, 4, 0, nt), c);
      c = wmma_f16(x1, b_frag(wf + OFF_W3, 4, 1, nt), c);
      store_tile_f16(T, 72, nt, c, S_M2, true);
    }
  }
  __builtin_amdgcn_wave_barrier();
  // L4: (16x64) @ (64x128) -> w0 (no activation), stays in accumulators
  v16h y0 = a_frag(T, 72, 0), y1 = a_frag(T, 72, 32);
  v8f c4[8];
#pragma unroll
  for (int nt = 0; nt < 8; ++nt) {
    v8f c = {};
    c = wmma_f16(y0, b_frag(wf + OFF_W4, 8, 0, nt), c);
    c = wmma_f16(y1, b_frag(wf + OFF_W4, 8, 1, nt), c);
    c4[nt] = c;
  }
  __builtin_amdgcn_wave_barrier();

  // gather h[sender], multiply by w0 * sh0, scatter-add into m[receiver]
  int snd[8], rcv[8]; float sh0[8];
#pragma unroll
  for (int r = 0; r < 8; ++r) {
    int el = hi * 8 + r;
    snd[r] = lsS[wave][el];
    rcv[r] = lsR[wave][el];
    sh0[r] = lsW[wave][el];
  }
#pragma unroll
  for (int nt = 0; nt < 8; ++nt) {
    int u = nt * 16 + M;
#pragma unroll
    for (int r = 0; r < 8; ++r) {
      float val = c4[nt][r] * S_W4 * sh0[r] * h[snd[r] * FF + u];
      unsafeAtomicAdd(&m[rcv[r] * FF + u], val);
    }
  }
}

// ---- post kernel: x1 = m@W_lin0 ; x2 = species-GEMM ; message = x2@W_out ----
__global__ __launch_bounds__(256) void k_post(const float* __restrict__ m,
                                              const int* __restrict__ species,
                                              const _Float16* __restrict__ wf,
                                              float* __restrict__ out) {
  __shared__ _Float16 lsA[8][16 * 136];
  __shared__ int lsp[8][16];
  int wave = threadIdx.x >> 5, lane = threadIdx.x & 31;
  int nb = (blockIdx.x * 8 + wave) * 16;
  if (nb >= NN) return;
  _Float16* A = lsA[wave];
  for (int i = lane; i < 16 * 128; i += 32) {
    int r = i >> 7, c = i & 127;
    A[r * 136 + c] = (_Float16)m[(nb + r) * FF + c];
  }
  if (lane < 16) lsp[wave][lane] = species[nb + lane];
  __builtin_amdgcn_wave_barrier();

  int M = lane & 15, hi = lane >> 4;
  // x1 = m @ W_lin0 * S_X1  -> back to LDS (f16)
  {
    v16h af[4];
#pragma unroll
    for (int kf = 0; kf < 4; ++kf) af[kf] = a_frag(A, 136, kf * 32);
    __builtin_amdgcn_wave_barrier();
#pragma unroll
    for (int nt = 0; nt < 8; ++nt) {
      v8f c = {};
#pragma unroll
      for (int kf = 0; kf < 4; ++kf)
        c = wmma_f16(af[kf], b_frag(wf + OFF_WLIN, 8, kf, nt), c);
      store_tile_f16(A, 136, nt, c, S_X1, false);
    }
  }
  __builtin_amdgcn_wave_barrier();
  // x2 = sum_s (x1 masked) @ W_skip_msg0[:,s,:] * S_X2 -> LDS
  {
    v16h af[4];
#pragma unroll
    for (int kf = 0; kf < 4; ++kf) af[kf] = a_frag(A, 136, kf * 32);
    __builtin_amdgcn_wave_barrier();
    int mysp = lsp[wave][M];
    v16h zv = {};
    v8f cs[8];
#pragma unroll
    for (int nt = 0; nt < 8; ++nt) { v8f z = {}; cs[nt] = z; }
    for (int s = 0; s < SS; ++s) {
      bool keep = (mysp == s);
      v16h az[4];
#pragma unroll
      for (int kf = 0; kf < 4; ++kf) az[kf] = keep ? af[kf] : zv;
      const _Float16* Ws = wf + OFF_WSM + s * (FF * FF);
#pragma unroll
      for (int nt = 0; nt < 8; ++nt)
#pragma unroll
        for (int kf = 0; kf < 4; ++kf)
          cs[nt] = wmma_f16(az[kf], b_frag(Ws, 8, kf, nt), cs[nt]);
    }
#pragma unroll
    for (int nt = 0; nt < 8; ++nt) store_tile_f16(A, 136, nt, cs[nt], S_X2, false);
  }
  __builtin_amdgcn_wave_barrier();
  // message = x2 @ W_out * S_OUT
  {
    v16h af[4];
#pragma unroll
    for (int kf = 0; kf < 4; ++kf) af[kf] = a_frag(A, 136, kf * 32);
#pragma unroll
    for (int nt = 0; nt < 8; ++nt) {
      v8f c = {};
#pragma unroll
      for (int kf = 0; kf < 4; ++kf)
        c = wmma_f16(af[kf], b_frag(wf + OFF_WOUT, 8, kf, nt), c);
#pragma unroll
      for (int r = 0; r < 8; ++r)
        out[(nb + hi * 8 + r) * FF + nt * 16 + M] = c[r] * S_OUT;
    }
  }
}

extern "C" void kernel_launch(void* const* d_in, const int* in_sizes, int n_in,
                              void* d_out, int out_size, void* d_ws, size_t ws_size,
                              hipStream_t stream) {
  const float* node_feats  = (const float*)d_in[1];
  const float* edge_attrs  = (const float*)d_in[2];
  const float* edge_feats  = (const float*)d_in[3];
  const int*   edge_index  = (const int*)d_in[4];
  const int*   species     = (const int*)d_in[5];
  const float* W_skip_node = (const float*)d_in[6];
  const float* W_up        = (const float*)d_in[7];
  const float* mlp_w1      = (const float*)d_in[8];
  const float* mlp_w2      = (const float*)d_in[9];
  const float* mlp_w3      = (const float*)d_in[10];
  const float* mlp_w4      = (const float*)d_in[11];
  const float* W_lin       = (const float*)d_in[12];
  const float* W_skip_msg  = (const float*)d_in[13];
  const float* W_out       = (const float*)d_in[14];

  float* h  = (float*)d_ws;                          // N*F f32
  float* mb = h + (size_t)NN * FF;                   // N*F f32 (scatter target)
  _Float16* wf = (_Float16*)(mb + (size_t)NN * FF);  // swizzled f16 weights

  float* msg_out  = (float*)d_out;
  float* skip_out = msg_out + (size_t)NN * FF;

  k_prep<<<512, 256, 0, stream>>>(W_up, mlp_w1, mlp_w2, mlp_w3, mlp_w4,
                                  W_lin, W_out, W_skip_node, W_skip_msg, wf);
  k_zero<<<1024, 256, 0, stream>>>(mb, NN * FF);
  int node_blocks = ((NN / 16) + 7) / 8;             // 79
  k_node<<<node_blocks, 256, 0, stream>>>(node_feats, species, wf, h, skip_out);
  int edge_blocks = EE / (16 * 8);                   // 1250
  k_edge<<<edge_blocks, 256, 0, stream>>>(edge_feats, edge_attrs, edge_index,
                                          h, wf, mb);
  k_post<<<node_blocks, 256, 0, stream>>>(mb, species, wf, msg_out);
}
